// Attention_60249801228345
// MI455X (gfx1250) — matrix operand order repaired and checked by reading, not compiled, not run on hardware
//
#include <hip/hip_runtime.h>
#include <hip/hip_bf16.h>

typedef __attribute__((ext_vector_type(16))) _Float16 v16h;
typedef __attribute__((ext_vector_type(8)))  _Float16 v8h;
typedef __attribute__((ext_vector_type(2)))  _Float16 v2h;
typedef __attribute__((ext_vector_type(8)))  float    v8f;

#define DIMX   1024
#define HEADS  16
#define DHEAD  64
#define SEQ    4096
#define BATCH  2
#define ROWS   8192      // BATCH*SEQ
#define QKVC   3072      // 3*DIMX
#define WIN    256

__device__ __forceinline__ v8f vzero8() {
  v8f z = {0.f,0.f,0.f,0.f,0.f,0.f,0.f,0.f};
  return z;
}

// A-matrix tile 16x32 f16: lane=row(l&15), khalf=l>>4.
// elements 0..7  = K (kbase + 8*kh   .. +7)
// elements 8..15 = K (kbase + 16+8*kh.. +7)
// caller passes p = base_of_row + kbase + 8*kh
__device__ __forceinline__ v16h load_a_tile(const _Float16* p) {
  v8h lo = *(const v8h*)(p);
  v8h hi = *(const v8h*)(p + 16);
  return __builtin_shufflevector(lo, hi, 0,1,2,3,4,5,6,7,8,9,10,11,12,13,14,15);
}

__device__ __forceinline__ v16h make16(v8h lo, v8h hi) {
  return __builtin_shufflevector(lo, hi, 0,1,2,3,4,5,6,7,8,9,10,11,12,13,14,15);
}

__device__ __forceinline__ v8f wmma_f16(v16h a, v16h b, v8f c) {
  // (neg_a, A, neg_b, B, c_mod, C, reuse_a, reuse_b)
  return __builtin_amdgcn_wmma_f32_16x16x32_f16(false, a, false, b, (short)0, c, false, false);
}

// CDNA5 async global->LDS copy (tracked by ASYNCcnt).  VDST = LDS byte
// address (low 32 bits of flat shared pointer), VADDR = 64-bit global addr.
__device__ __forceinline__ void async_ld_b128(const void* lds_ptr, const void* gptr) {
  unsigned lds_off = (unsigned)(size_t)lds_ptr;
  unsigned long long ga = (unsigned long long)(size_t)gptr;
  asm volatile("global_load_async_to_lds_b128 %0, %1, off"
               :: "v"(lds_off), "v"(ga) : "memory");
}

__device__ __forceinline__ void wait_async_keep2() {
  asm volatile("s_wait_asynccnt 0x2" ::: "memory");
}
__device__ __forceinline__ void wait_async_all() {
  asm volatile("s_wait_asynccnt 0x0" ::: "memory");
}

// issue one 8KB weight chunk cooperatively: this thread's 32B slice (2 b128)
__device__ __forceinline__ void issue_chunk(const _Float16* gb, _Float16* lb) {
  async_ld_b128(lb, gb);
  async_ld_b128(lb + 8, gb + 8);
}

// one k-chunk of the 128-wide block tile: 1 A tile x 8 B tiles -> 8 WMMAs
__device__ __forceinline__ void compute_chunk(const _Float16* pa_c, const _Float16* lb,
                                              int n, int kh, v8f* acc) {
  v16h a = load_a_tile(pa_c);
#pragma unroll
  for (int j = 0; j < 8; ++j) {
    v16h b = make16(*(const v8h*)(lb + (j * 16 + n) * 32 + 8 * kh), *(const v8h*)(lb + (j * 16 + n) * 32 + 16 + 8 * kh));
    acc[j] = wmma_f16(a, b, acc[j]);
  }
}

// ---------------------------------------------------------------------------
// Weight transpose + f32->f16 convert: W[k][n] -> Wt[n][k]
// ---------------------------------------------------------------------------
__global__ void k_conv_wqkv(const float* __restrict__ W, _Float16* __restrict__ Wt) {
  int i = blockIdx.x * 256 + threadIdx.x;
  if (i >= DIMX * QKVC) return;
  int k = i / QKVC;
  int n = i - k * QKVC;
  Wt[(size_t)n * DIMX + k] = (_Float16)W[i];
}

__global__ void k_conv_wout(const float* __restrict__ W, _Float16* __restrict__ Wt) {
  int i = blockIdx.x * 256 + threadIdx.x;
  if (i >= DIMX * DIMX) return;
  int k = i >> 10;
  int n = i & 1023;
  Wt[(size_t)n * DIMX + k] = (_Float16)W[i];
}

// ---------------------------------------------------------------------------
// RMSNorm (writes f16 xn) + head gates sigmoid(xn@Wg + bg)
// one block (256 thr) per row
// ---------------------------------------------------------------------------
__global__ void k_rms_gates(const float* __restrict__ X, const float* __restrict__ gamma,
                            const float* __restrict__ Wg, const float* __restrict__ bg,
                            _Float16* __restrict__ XN, float* __restrict__ Gs) {
  __shared__ float red[256];
  __shared__ float gsh[256 * HEADS];
  int row = blockIdx.x;
  int tid = threadIdx.x;
  const float* xr = X + (size_t)row * DIMX;
  float ss = 0.f;
  for (int k = tid; k < DIMX; k += 256) { float v = xr[k]; ss += v * v; }
  red[tid] = ss;
  __syncthreads();
  for (int s = 128; s > 0; s >>= 1) {
    if (tid < s) red[tid] += red[tid + s];
    __syncthreads();
  }
  float fac = 32.0f / fmaxf(sqrtf(red[0]), 1e-12f);   // sqrt(1024)=32
  float gp[HEADS];
#pragma unroll
  for (int h = 0; h < HEADS; ++h) gp[h] = 0.f;
  for (int k = tid; k < DIMX; k += 256) {
    float xv = xr[k] * fac * gamma[k];
    XN[(size_t)row * DIMX + k] = (_Float16)xv;
    const float* wgr = Wg + (size_t)k * HEADS;
#pragma unroll
    for (int h = 0; h < HEADS; ++h) gp[h] += xv * wgr[h];
  }
#pragma unroll
  for (int h = 0; h < HEADS; ++h) gsh[tid * HEADS + h] = gp[h];
  __syncthreads();
  if (tid < HEADS) {
    float acc = bg[tid];
    for (int t = 0; t < 256; ++t) acc += gsh[t * HEADS + tid];
    Gs[(size_t)row * HEADS + tid] = 1.f / (1.f + __expf(-acc));
  }
}

// ---------------------------------------------------------------------------
// QKV GEMM: [ROWS,DIMX] f16 @ Wt[QKVC,DIMX] f16 -> scatter Q,K [b,h,S,d] and
// Vt (transposed) [b,h,d,S], all f16.
// Block (8 waves) computes a 128x128 tile; the shared 128x32 weight chunk
// (8KB) is async-staged into LDS, double-buffered, k-loop unrolled by 2 so
// each buffer phase is explicit (no buffer-select phi on accumulators).
// ---------------------------------------------------------------------------
__global__ void k_gemm_qkv(const _Float16* __restrict__ A, const _Float16* __restrict__ Bt,
                           _Float16* __restrict__ Qf, _Float16* __restrict__ Kf,
                           _Float16* __restrict__ Vt) {
  __shared__ _Float16 ldsB[2][128 * 32];         // 8 KB per buffer
  int tid  = threadIdx.x;
  int wave = tid >> 5;
  int lane = tid & 31;
  int rowBlk  = blockIdx.x / 24;                 // 64 row blocks of 128
  int colTile = blockIdx.x % 24;                 // 24 col tiles of 128
  int rowTile = rowBlk * 8 + wave;               // this wave's 16-row tile
  int n  = lane & 15;
  int kh = lane >> 4;
  const _Float16* pa = A + (size_t)(rowTile * 16 + n) * DIMX + 8 * kh;
  // cooperative B loader: 256 threads x 32B = 8KB chunk (128 n-rows x 32 k)
  int nloc = tid >> 1;                           // 0..127
  int pp   = tid & 1;                            // 32B half of the 64B row
  const _Float16* gb = Bt + (size_t)(colTile * 128 + nloc) * DIMX + pp * 16;
  _Float16* lb0 = &ldsB[0][nloc * 32 + pp * 16];
  _Float16* lb1 = &ldsB[1][nloc * 32 + pp * 16];

  v8f acc[8];
#pragma unroll
  for (int j = 0; j < 8; ++j) acc[j] = vzero8();

  issue_chunk(gb, lb0);                          // chunk 0
  issue_chunk(gb + 32, lb1);                     // chunk 1
  for (int c = 0; c < 32; c += 2) {
    // ---- phase A: buffer 0, chunk c ----
    wait_async_keep2();                          // chunk c landed, c+1 in flight
    __syncthreads();
    compute_chunk(pa + c * 32, &ldsB[0][0], n, kh, acc);
    __syncthreads();                             // buf0 free
    if (c + 2 < 32) {
      issue_chunk(gb + (c + 2) * 32, lb0);
      wait_async_keep2();                        // chunk c+1 landed
    } else {
      wait_async_all();
    }
    // ---- phase B: buffer 1, chunk c+1 ----
    __syncthreads();
    compute_chunk(pa + (c + 1) * 32, &ldsB[1][0], n, kh, acc);
    __syncthreads();                             // buf1 free
    if (c + 3 < 32)
      issue_chunk(gb + (c + 3) * 32, lb1);
  }

#pragma unroll
  for (int j = 0; j < 8; ++j) {
    int col = colTile * 128 + j * 16 + n;
    int sel = col >> 10;          // 0=q 1=k 2=v
    int rem = col & 1023;
    int hh  = rem >> 6;
    int d   = rem & 63;
#pragma unroll
    for (int r = 0; r < 8; ++r) {
      int m  = rowTile * 16 + r + 8 * kh;
      int b2 = m >> 12;           // batch
      int s  = m & 4095;          // seq pos
      _Float16 v = (_Float16)acc[j][r];
      if (sel == 0)
        Qf[((size_t)(b2 * HEADS + hh) * SEQ + s) * DHEAD + d] = v;
      else if (sel == 1)
        Kf[((size_t)(b2 * HEADS + hh) * SEQ + s) * DHEAD + d] = v;
      else
        Vt[((size_t)(b2 * HEADS + hh) * DHEAD + d) * SEQ + s] = v;
    }
  }
}

// ---------------------------------------------------------------------------
// Sliding-window attention + gating.  One wave -> one (b,h, 16-query tile).
// Computes S^T = K*Q^T (scores in key-major C tiles), online softmax,
// O^T = V^T * P^T; epilogue applies 1/sum and sigmoid gate, writes f16
// AO[b,S,h*64+d].
// ---------------------------------------------------------------------------
__global__ void k_attn(const _Float16* __restrict__ Qf, const _Float16* __restrict__ Kf,
                       const _Float16* __restrict__ Vt, const float* __restrict__ Gs,
                       _Float16* __restrict__ AO) {
  int gw   = (blockIdx.x * blockDim.x + threadIdx.x) >> 5;
  int lane = threadIdx.x & 31;
  int qt = gw & 255;
  int h  = (gw >> 8) & 15;
  int bb = gw >> 12;
  int qs = qt << 4;
  int n  = lane & 15;
  int kh = lane >> 4;
  const _Float16* Qp = Qf + (size_t)(bb * HEADS + h) * SEQ * DHEAD;
  const _Float16* Kp = Kf + (size_t)(bb * HEADS + h) * SEQ * DHEAD;
  const _Float16* Vp = Vt + (size_t)(bb * HEADS + h) * DHEAD * SEQ;

  // Q^T B-tiles (32 d x 16 queries), lane=query col
  v16h bq0 = make16(*(const v8h*)(Qp + (size_t)(qs + n) * DHEAD + 8 * kh), *(const v8h*)(Qp + (size_t)(qs + n) * DHEAD + 16 + 8 * kh));
  v16h bq1 = make16(*(const v8h*)(Qp + (size_t)(qs + n) * DHEAD + 32 + 8 * kh), *(const v8h*)(Qp + (size_t)(qs + n) * DHEAD + 48 + 8 * kh));

  v8f o[4];
#pragma unroll
  for (int t = 0; t < 4; ++t) o[t] = vzero8();
  float mrun = -1e30f, srun = 0.f;
  int iq = qs + n;                      // this lane's query index

  for (int c = 0; c < 9; ++c) {         // 9 chunks of 32 keys covering [qs-256, qs+32)
    int k0 = qs - 256 + 32 * c;
    int k1 = k0 + 16;
    bool val0 = (k0 >= 0) && (k0 <= qs);
    bool val1 = (k1 >= 0) && (k1 <= qs);
    if (!val0 && !val1) continue;       // wave-uniform

    // prefetch next chunk's K rows into cache while computing this one
    int kpf = k0 + 32;
    if (kpf >= 0 && kpf <= qs)
      __builtin_prefetch(Kp + (size_t)(kpf + n) * DHEAD + 8 * kh, 0, 0);

    v8f sc0 = vzero8(), sc1 = vzero8();
    if (val0) {
      v16h a0 = load_a_tile(Kp + (size_t)(k0 + n) * DHEAD + 8 * kh);
      v16h a1 = load_a_tile(Kp + (size_t)(k0 + n) * DHEAD + 32 + 8 * kh);
      sc0 = wmma_f16(a0, bq0, sc0);
      sc0 = wmma_f16(a1, bq1, sc0);
    }
    if (val1) {
      v16h a0 = load_a_tile(Kp + (size_t)(k1 + n) * DHEAD + 8 * kh);
      v16h a1 = load_a_tile(Kp + (size_t)(k1 + n) * DHEAD + 32 + 8 * kh);
      sc1 = wmma_f16(a0, bq0, sc1);
      sc1 = wmma_f16(a1, bq1, sc1);
    }

    float x0[8], x1[8];
    float cmax = -1e30f;
#pragma unroll
    for (int r = 0; r < 8; ++r) {
      int j0 = k0 + r + 8 * kh;
      int j1 = k1 + r + 8 * kh;
      x0[r] = (val0 && j0 <= iq && (iq - j0) <= WIN) ? sc0[r] * 0.125f : -1e30f;
      x1[r] = (val1 && j1 <= iq && (iq - j1) <= WIN) ? sc1[r] * 0.125f : -1e30f;
      cmax = fmaxf(cmax, fmaxf(x0[r], x1[r]));
    }
    cmax = fmaxf(cmax, __shfl_xor(cmax, 16, 32));     // per-query chunk max
    float mnew  = fmaxf(mrun, cmax);
    float alpha = __expf(mrun - mnew);
    float psum = 0.f;
    float p0[8], p1[8];
#pragma unroll
    for (int r = 0; r < 8; ++r) {
      p0[r] = __expf(x0[r] - mnew);
      p1[r] = __expf(x1[r] - mnew);
      psum += p0[r] + p1[r];
    }
    psum += __shfl_xor(psum, 16, 32);
    srun = srun * alpha + psum;
#pragma unroll
    for (int t = 0; t < 4; ++t) o[t] *= alpha;

    // Build P^T as B tile (32 keys x 16 queries): elements 0..7 = p0 (keys k0+8*kh..+7),
    // elements 8..15 = p1 (keys k1+8*kh..+7); own values only, shfl_xor results unused.
    v16h bp;
#pragma unroll
    for (int r = 0; r < 8; ++r) {
      float q0 = __shfl_xor(p0[r], 16, 32);
      float q1 = __shfl_xor(p1[r], 16, 32);
      bp[r]     = (_Float16)p0[r];
      bp[r + 8] = (_Float16)p1[r];
    }

    // V^T A-tiles (16 d-rows x 32 keys) from transposed V; invalid half keys
    // are multiplied by zero P, so redirect their loads to the valid half.
    int ko1 = (val0 ? k0 : k1) + 8 * kh;
    int ko2 = (val1 ? k1 : k0) + 8 * kh;
#pragma unroll
    for (int t = 0; t < 4; ++t) {
      const _Float16* pr = Vp + (size_t)(t * 16 + n) * SEQ;
      v8h lo = *(const v8h*)(pr + ko1);
      v8h hi = *(const v8h*)(pr + ko2);
      o[t] = wmma_f16(make16(lo, hi), bp, o[t]);
    }
    mrun = mnew;
  }

  float g = (1.0f / srun) * Gs[(size_t)(bb * SEQ + iq) * HEADS + h];
  _Float16* aor = AO + (size_t)(bb * SEQ + iq) * (HEADS * DHEAD) + h * DHEAD;
#pragma unroll
  for (int t = 0; t < 4; ++t) {
#pragma unroll
    for (int r = 0; r < 8; r += 2) {
      int d = t * 16 + r + 8 * kh;
      v2h pr2 = { (_Float16)(o[t][r] * g), (_Float16)(o[t][r + 1] * g) };
      *(v2h*)(aor + d) = pr2;
    }
  }
}

// ---------------------------------------------------------------------------
// Output GEMM: AO[ROWS,DIMX] f16 @ WtOut[DIMX,DIMX] f16 -> f32 d_out
// Same 128x128 block-cooperative LDS staging as k_gemm_qkv.
// ---------------------------------------------------------------------------
__global__ void k_gemm_out(const _Float16* __restrict__ A, const _Float16* __restrict__ Bt,
                           float* __restrict__ Out) {
  __shared__ _Float16 ldsB[2][128 * 32];
  int tid  = threadIdx.x;
  int wave = tid >> 5;
  int lane = tid & 31;
  int rowBlk  = blockIdx.x >> 3;                 // 64 row blocks of 128
  int colTile = blockIdx.x & 7;                  // 8 col tiles of 128
  int rowTile = rowBlk * 8 + wave;
  int n  = lane & 15;
  int kh = lane >> 4;
  const _Float16* pa = A + (size_t)(rowTile * 16 + n) * DIMX + 8 * kh;
  int nloc = tid >> 1;
  int pp   = tid & 1;
  const _Float16* gb = Bt + (size_t)(colTile * 128 + nloc) * DIMX + pp * 16;
  _Float16* lb0 = &ldsB[0][nloc * 32 + pp * 16];
  _Float16* lb1 = &ldsB[1][nloc * 32 + pp * 16];

  v8f acc[8];
#pragma unroll
  for (int j = 0; j < 8; ++j) acc[j] = vzero8();

  issue_chunk(gb, lb0);
  issue_chunk(gb + 32, lb1);
  for (int c = 0; c < 32; c += 2) {
    wait_async_keep2();
    __syncthreads();
    compute_chunk(pa + c * 32, &ldsB[0][0], n, kh, acc);
    __syncthreads();
    if (c + 2 < 32) {
      issue_chunk(gb + (c + 2) * 32, lb0);
      wait_async_keep2();
    } else {
      wait_async_all();
    }
    __syncthreads();
    compute_chunk(pa + (c + 1) * 32, &ldsB[1][0], n, kh, acc);
    __syncthreads();
    if (c + 3 < 32)
      issue_chunk(gb + (c + 3) * 32, lb1);
  }

#pragma unroll
  for (int j = 0; j < 8; ++j) {
    int col = colTile * 128 + j * 16 + n;
#pragma unroll
    for (int r = 0; r < 8; ++r) {
      int m = rowTile * 16 + r + 8 * kh;
      Out[(size_t)m * DIMX + col] = acc[j][r];
    }
  }
}

// ---------------------------------------------------------------------------
extern "C" void kernel_launch(void* const* d_in, const int* in_sizes, int n_in,
                              void* d_out, int out_size, void* d_ws, size_t ws_size,
                              hipStream_t stream) {
  (void)in_sizes; (void)n_in; (void)out_size; (void)ws_size;
  const float* X     = (const float*)d_in[0];
  const float* gamma = (const float*)d_in[1];
  const float* Wqkv  = (const float*)d_in[2];
  const float* Wg    = (const float*)d_in[3];
  const float* bg    = (const float*)d_in[4];
  const float* Wout  = (const float*)d_in[5];
  float* Out = (float*)d_out;

  char* ws = (char*)d_ws;
  size_t off = 0;
  auto take = [&](size_t bytes) {
    size_t o = off;
    off += (bytes + 255) & ~(size_t)255;
    return o;
  };
  _Float16* XN  = (_Float16*)(ws + take((size_t)ROWS * DIMX * 2));   // reused as AO
  _Float16* WTQ = (_Float16*)(ws + take((size_t)QKVC * DIMX * 2));
  _Float16* WTO = (_Float16*)(ws + take((size_t)DIMX * DIMX * 2));
  _Float16* Qf  = (_Float16*)(ws + take((size_t)BATCH * HEADS * SEQ * DHEAD * 2));
  _Float16* Kf  = (_Float16*)(ws + take((size_t)BATCH * HEADS * SEQ * DHEAD * 2));
  _Float16* Vt  = (_Float16*)(ws + take((size_t)BATCH * HEADS * SEQ * DHEAD * 2));
  float*    Gs  = (float*)   (ws + take((size_t)ROWS * HEADS * 4));
  _Float16* AO  = XN;   // XN is dead after k_gemm_qkv; reuse for attention out

  k_conv_wqkv<<<(DIMX * QKVC + 255) / 256, 256, 0, stream>>>(Wqkv, WTQ);
  k_conv_wout<<<(DIMX * DIMX + 255) / 256, 256, 0, stream>>>(Wout, WTO);
  k_rms_gates<<<ROWS, 256, 0, stream>>>(X, gamma, Wg, bg, XN, Gs);
  // 64 row blocks * 24 col tiles, 128x128 tile per block
  k_gemm_qkv<<<64 * 24, 256, 0, stream>>>(XN, WTQ, Qf, Kf, Vt);
  // 2*16*256 = 8192 waves
  k_attn<<<8192 / 8, 256, 0, stream>>>(Qf, Kf, Vt, Gs, AO);
  // 64 row blocks * 8 col tiles
  k_gemm_out<<<64 * 8, 256, 0, stream>>>(AO, WTO, Out);
}
